// IPEXTransformerAtten_16801912062547
// MI455X (gfx1250) — matrix operand order repaired and checked by reading, not compiled, not run on hardware
//
#include <hip/hip_runtime.h>
#include <hip/hip_bf16.h>

// Problem constants (match reference)
#define BB 2
#define SS 2048
#define EE 4096
#define HH 32
#define DD 128

typedef __attribute__((ext_vector_type(16))) __bf16 v16bf;
typedef __attribute__((ext_vector_type(8)))  float  v8f;

union Frag { v16bf v; uint4 q[2]; };

__device__ __forceinline__ v8f zero8() {
  v8f z = {0.f,0.f,0.f,0.f,0.f,0.f,0.f,0.f};
  return z;
}

// ---------------------------------------------------------------------------
// CDNA5 async global->LDS copy (ASYNCcnt-tracked, no VGPR round trip).
// Low 32 bits of a generic pointer to LDS are the LDS byte address (ISA 10.2).
// ---------------------------------------------------------------------------
__device__ __forceinline__ void async_b128(void* lds, const void* g) {
  unsigned off = (unsigned)(uintptr_t)lds;
  asm volatile("global_load_async_to_lds_b128 %0, %1, off"
               :: "v"(off), "v"(g) : "memory");
}
__device__ __forceinline__ void wait_async0() {
  asm volatile("s_wait_asynccnt 0x0" ::: "memory");
}

// ---------------------------------------------------------------------------
// 16-lane butterfly reductions as pure VALU (v_permlane16_b32), replacing
// ds_bpermute-based __shfl_xor. Both 16-lane rows reduce independently.
// ---------------------------------------------------------------------------
__device__ __forceinline__ float red16_max(float x) {
  union { float f; unsigned u; } a, b;
  a.f = x;
  b.u = __builtin_amdgcn_permlane16(a.u, a.u, 0x67452301u, 0xEFCDAB89u, false, false); // ^1
  a.f = fmaxf(a.f, b.f);
  b.u = __builtin_amdgcn_permlane16(a.u, a.u, 0x54761032u, 0xDCFE98BAu, false, false); // ^2
  a.f = fmaxf(a.f, b.f);
  b.u = __builtin_amdgcn_permlane16(a.u, a.u, 0x32107654u, 0xBA98FEDCu, false, false); // ^4
  a.f = fmaxf(a.f, b.f);
  b.u = __builtin_amdgcn_permlane16(a.u, a.u, 0xFEDCBA98u, 0x76543210u, false, false); // ^8
  a.f = fmaxf(a.f, b.f);
  return a.f;
}
__device__ __forceinline__ float red16_sum(float x) {
  union { float f; unsigned u; } a, b;
  a.f = x;
  b.u = __builtin_amdgcn_permlane16(a.u, a.u, 0x67452301u, 0xEFCDAB89u, false, false);
  a.f += b.f;
  b.u = __builtin_amdgcn_permlane16(a.u, a.u, 0x54761032u, 0xDCFE98BAu, false, false);
  a.f += b.f;
  b.u = __builtin_amdgcn_permlane16(a.u, a.u, 0x32107654u, 0xBA98FEDCu, false, false);
  a.f += b.f;
  b.u = __builtin_amdgcn_permlane16(a.u, a.u, 0xFEDCBA98u, 0x76543210u, false, false);
  a.f += b.f;
  return a.f;
}

// ---------------------------------------------------------------------------
// f32 -> bf16 elementwise convert (vectorized)
// ---------------------------------------------------------------------------
__global__ void cvt_f32_bf16(const float* __restrict__ in, __bf16* __restrict__ out, int n4) {
  int i = blockIdx.x * blockDim.x + threadIdx.x;
  if (i < n4) {
    float4 f = ((const float4*)in)[i];
    union { __bf16 h[4]; uint2 u; } pk;
    pk.h[0] = (__bf16)f.x; pk.h[1] = (__bf16)f.y;
    pk.h[2] = (__bf16)f.z; pk.h[3] = (__bf16)f.w;
    *(uint2*)&out[4 * (size_t)i] = pk.u;
  }
}

// ---------------------------------------------------------------------------
// f32 [R][Cn] -> bf16 transposed [Cn][R] (so GEMM B operand is K-contiguous)
// ---------------------------------------------------------------------------
__global__ void transpose_cvt(const float* __restrict__ in, __bf16* __restrict__ out,
                              int R, int Cn) {
  __shared__ float t[32][33];
  int c0 = blockIdx.x * 32, r0 = blockIdx.y * 32;
  int tx = threadIdx.x, ty = threadIdx.y;
#pragma unroll
  for (int i = 0; i < 4; ++i)
    t[ty + 8 * i][tx] = in[(size_t)(r0 + ty + 8 * i) * Cn + c0 + tx];
  __syncthreads();
#pragma unroll
  for (int i = 0; i < 4; ++i)
    out[(size_t)(c0 + ty + 8 * i) * R + r0 + tx] = (__bf16)t[tx][ty + 8 * i];
}

// ---------------------------------------------------------------------------
// bf16 WMMA GEMM:  C[M][N] = A[M][K] * Bt[N][K]^T + bias
// 256 threads = 8 waves; block tile 128(M) x 64(N); wave tile 32x32 = 2x2 WMMA
// Double-buffered LDS fed by global_load_async_to_lds_b128 (ASYNCcnt pipeline):
//   wait(current) -> barrier -> issue(next, other buf) -> compute(current)
// ---------------------------------------------------------------------------
template <typename OutT>
__global__ __launch_bounds__(256)
void gemm_bf16_wmma(const __bf16* __restrict__ A, const __bf16* __restrict__ Bt,
                    const float* __restrict__ bias, OutT* __restrict__ C,
                    int M, int N, int K) {
  __shared__ __bf16 ldsA[2][128 * 40];  // 128 rows x 32 K, padded to 40 (16B aligned)
  __shared__ __bf16 ldsB[2][64 * 40];   // 64 rows (N) x 32 K, padded

  const int tid  = threadIdx.x;
  const int lane = tid & 31;
  const int wv   = tid >> 5;
  const int l16  = lane & 15;
  const bool lo  = lane < 16;
  const int wm   = wv & 3, wn = wv >> 2;        // 4 x 2 wave grid
  const int m0   = blockIdx.y * 128, n0 = blockIdx.x * 64;

  const int arow = tid & 127, apart = tid >> 7; // each thread: 32B of one A row
  const int brow = tid & 63,  bpart = tid >> 6; // each thread: 16B of one B row

  const __bf16* agBase = A  + (size_t)(m0 + arow) * K + apart * 16;
  const __bf16* bgBase = Bt + (size_t)(n0 + brow) * K + bpart * 8;
  const int aoff = arow * 40 + apart * 16;
  const int boff = brow * 40 + bpart * 8;

  v8f acc[2][2] = {{zero8(), zero8()}, {zero8(), zero8()}};

  // prologue: async-stage K-panel 0 into buffer 0
  async_b128(&ldsA[0][aoff],     agBase);
  async_b128(&ldsA[0][aoff + 8], agBase + 8);
  async_b128(&ldsB[0][boff],     bgBase);

  int buf = 0;
  for (int kk = 0; kk < K; kk += 32) {
    wait_async0();       // this wave's tile-kk loads have landed in LDS
    __syncthreads();     // all waves' tiles visible; all done reading buf^1

    if (kk + 32 < K) {   // overlap: stage next panel into the other buffer
      async_b128(&ldsA[buf ^ 1][aoff],     agBase + kk + 32);
      async_b128(&ldsA[buf ^ 1][aoff + 8], agBase + kk + 40);
      async_b128(&ldsB[buf ^ 1][boff],     bgBase + kk + 32);
    }

    Frag a[2], b[2];
#pragma unroll
    for (int mt = 0; mt < 2; ++mt) {
      // ISA 16-bit A layout: lane<16 -> K{0..7,16..23}; lane>=16 -> K{8..15,24..31}
      int base = (wm * 32 + mt * 16 + l16) * 40 + (lo ? 0 : 8);
      a[mt].q[0] = *(const uint4*)&ldsA[buf][base];
      a[mt].q[1] = *(const uint4*)&ldsA[buf][base + 16];
    }
#pragma unroll
    for (int nt = 0; nt < 2; ++nt) {
      // 16-bit B layout (matches A): lane<16 -> K{0..7,16..23}; lane>=16 -> K{8..15,24..31}
      int base = (wn * 32 + nt * 16 + l16) * 40 + (lo ? 0 : 8);
      b[nt].q[0] = *(const uint4*)&ldsB[buf][base];
      b[nt].q[1] = *(const uint4*)&ldsB[buf][base + 16];
    }
#pragma unroll
    for (int mt = 0; mt < 2; ++mt)
#pragma unroll
      for (int nt = 0; nt < 2; ++nt)
        acc[mt][nt] = __builtin_amdgcn_wmma_f32_16x16x32_bf16(
            false, a[mt].v, false, b[nt].v, (short)0, acc[mt][nt], false, false);

    buf ^= 1;
  }

  // C layout: lane n = l16; VGPR r -> row r (lanes 0-15) / r+8 (lanes 16-31)
#pragma unroll
  for (int mt = 0; mt < 2; ++mt)
#pragma unroll
    for (int nt = 0; nt < 2; ++nt) {
      int gn = n0 + wn * 32 + nt * 16 + l16;
      float bv = bias[gn];
#pragma unroll
      for (int r = 0; r < 8; ++r) {
        int gm = m0 + wm * 32 + mt * 16 + r + (lo ? 0 : 8);
        C[(size_t)gm * N + gn] = (OutT)(acc[mt][nt][r] + bv);
      }
    }
}

// ---------------------------------------------------------------------------
// RoPE + split: qkv bf16 [B,S,3E] -> Q,K bf16 [B,H,S,D], V transposed [B,H,D,S]
// ---------------------------------------------------------------------------
__global__ void rope_split(const __bf16* __restrict__ qkv, __bf16* __restrict__ Q,
                           __bf16* __restrict__ Ko, __bf16* __restrict__ Vt) {
  int s = blockIdx.x, h = blockIdx.y, b = blockIdx.z, d = threadIdx.x;
  size_t bh  = (size_t)b * HH + h;
  size_t inb = ((size_t)(b * SS + s)) * (3 * EE) + h * DD;
  float x   = (float)qkv[inb + d];
  float xk  = (float)qkv[inb + EE + d];
  float xo  = (float)qkv[inb + (d ^ 64)];
  float xko = (float)qkv[inb + EE + (d ^ 64)];
  __bf16 v  = qkv[inb + 2 * EE + d];
  int j = d & 63;
  float inv = __expf(-0.14391156f * (float)j);  // 10000^(-j/64)
  float ang = (float)s * inv;
  float sn, cs;
  __sincosf(ang, &sn, &cs);
  float qr, kr;
  if (d < 64) { qr = x * cs - xo * sn;  kr = xk * cs - xko * sn; }
  else        { qr = x * cs + xo * sn;  kr = xk * cs + xko * sn; }
  Q [(bh * SS + s) * DD + d] = (__bf16)qr;
  Ko[(bh * SS + s) * DD + d] = (__bf16)kr;
  Vt[(bh * DD + d) * SS + s] = v;
}

// ---------------------------------------------------------------------------
// Flash attention (causal), per-(b,h): 4 waves x 16 query rows per block.
// Per 32-key step: 8 WMMA (QK^T) + online softmax (permlane16 reductions)
// + LDS P-transpose + 8 WMMA (PV)
// ---------------------------------------------------------------------------
__global__ __launch_bounds__(128)
void flash_attn_wmma(const __bf16* __restrict__ Q, const __bf16* __restrict__ K,
                     const __bf16* __restrict__ Vt, __bf16* __restrict__ ctx) {
  __shared__ __bf16 ldsP[4][16 * 40];   // per-wave P scratch (16 x 32, padded)
  const int tid  = threadIdx.x;
  const int lane = tid & 31, w = tid >> 5;
  const int l16  = lane & 15;
  const bool lo  = lane < 16;
  const int hi8  = lo ? 0 : 8;
  const int b = blockIdx.z, h = blockIdx.y;
  const size_t bh = (size_t)b * HH + h;
  const __bf16* Qh = Q  + bh * SS * DD;
  const __bf16* Kh = K  + bh * SS * DD;
  const __bf16* Vh = Vt + bh * DD * SS;
  const int q0 = blockIdx.x * 64 + w * 16;
  const float scale = 0.08838834764831845f;   // 1/sqrt(128)

  // Q tile 16x128 resident as 4 A-fragments
  Frag aq[4];
  {
    int row = q0 + l16;
#pragma unroll
    for (int c = 0; c < 4; ++c) {
      const __bf16* p = Qh + (size_t)row * DD + c * 32 + (lo ? 0 : 8);
      aq[c].q[0] = *(const uint4*)p;
      aq[c].q[1] = *(const uint4*)(p + 16);
    }
  }

  v8f O[8];
  float mrow[8], lrow[8];
#pragma unroll
  for (int c = 0; c < 8; ++c) O[c] = zero8();
#pragma unroll
  for (int r = 0; r < 8; ++r) { mrow[r] = -3.0e38f; lrow[r] = 0.f; }

  __bf16* Pw = ldsP[w];

  for (int kk = 0; kk < q0 + 16; kk += 32) {
    v8f s0 = zero8(), s1 = zero8();
#pragma unroll
    for (int c = 0; c < 4; ++c) {
      Frag bk;
      const __bf16* p = Kh + (size_t)(kk + l16) * DD + c * 32 + (lo ? 0 : 8);
      bk.q[0] = *(const uint4*)p;
      bk.q[1] = *(const uint4*)(p + 16);
      s0 = __builtin_amdgcn_wmma_f32_16x16x32_bf16(false, aq[c].v, false, bk.v,
                                                   (short)0, s0, false, false);
    }
#pragma unroll
    for (int c = 0; c < 4; ++c) {
      Frag bk;
      const __bf16* p = Kh + (size_t)(kk + 16 + l16) * DD + c * 32 + (lo ? 0 : 8);
      bk.q[0] = *(const uint4*)p;
      bk.q[1] = *(const uint4*)(p + 16);
      s1 = __builtin_amdgcn_wmma_f32_16x16x32_bf16(false, aq[c].v, false, bk.v,
                                                   (short)0, s1, false, false);
    }

    float alpha[8];
#pragma unroll
    for (int r = 0; r < 8; ++r) {
      int qi  = q0 + r + hi8;
      int k0i = kk + l16, k1i = kk + 16 + l16;
      float f0 = s0[r] * scale + (k0i > qi ? -66504.0f : 0.0f);
      float f1 = s1[r] * scale + (k1i > qi ? -66504.0f : 0.0f);
      float t  = red16_max(fmaxf(f0, f1));
      float nm = fmaxf(mrow[r], t);
      float al = __expf(mrow[r] - nm);
      mrow[r]  = nm;
      alpha[r] = al;
      float p0 = __expf(f0 - nm);
      float p1 = __expf(f1 - nm);
      float rs = red16_sum(p0 + p1);
      lrow[r] = lrow[r] * al + rs;
      // write P (C-layout -> row-major in LDS) for A-layout reload
      Pw[(r + hi8) * 40 + l16]      = (__bf16)p0;
      Pw[(r + hi8) * 40 + 16 + l16] = (__bf16)p1;
    }

    asm volatile("s_wait_dscnt 0" ::: "memory");

    Frag ap;
    {
      int base = l16 * 40 + (lo ? 0 : 8);
      ap.q[0] = *(const uint4*)&Pw[base];
      ap.q[1] = *(const uint4*)&Pw[base + 16];
    }

#pragma unroll
    for (int c = 0; c < 8; ++c) {
#pragma unroll
      for (int r = 0; r < 8; ++r) O[c][r] *= alpha[r];
      Frag bv;
      const __bf16* p = Vh + (size_t)(c * 16 + l16) * SS + kk + (lo ? 0 : 8);
      bv.q[0] = *(const uint4*)p;
      bv.q[1] = *(const uint4*)(p + 16);
      O[c] = __builtin_amdgcn_wmma_f32_16x16x32_bf16(false, ap.v, false, bv.v,
                                                     (short)0, O[c], false, false);
    }
  }

#pragma unroll
  for (int c = 0; c < 8; ++c) {
#pragma unroll
    for (int r = 0; r < 8; ++r) {
      int qi = q0 + r + hi8;
      size_t o = ((size_t)(b * SS + qi)) * EE + h * DD + c * 16 + l16;
      ctx[o] = (__bf16)(O[c][r] / lrow[r]);
    }
  }
}

// ---------------------------------------------------------------------------
extern "C" void kernel_launch(void* const* d_in, const int* in_sizes, int n_in,
                              void* d_out, int out_size, void* d_ws, size_t ws_size,
                              hipStream_t stream) {
  const float* hidden = (const float*)d_in[0];
  const float* qkv_w  = (const float*)d_in[1];
  const float* qkv_b  = (const float*)d_in[2];
  const float* out_w  = (const float*)d_in[3];
  const float* out_b  = (const float*)d_in[4];
  float* out = (float*)d_out;

  // workspace layout (total = 384 MB)
  char* ws = (char*)d_ws;
  __bf16* qkv_wt = (__bf16*)ws; ws += (size_t)3 * EE * EE * 2;     // 96 MB [3E][E]
  __bf16* out_wt = (__bf16*)ws; ws += (size_t)EE * EE * 2;         // 32 MB [E][E]
  __bf16* hid_bf = (__bf16*)ws; ws += (size_t)BB * SS * EE * 2;    // 32 MB
  __bf16* qkv_bf = (__bf16*)ws; ws += (size_t)BB * SS * 3 * EE * 2;// 96 MB
  __bf16* Qb     = (__bf16*)ws; ws += (size_t)BB * SS * EE * 2;    // 32 MB
  __bf16* Kb     = (__bf16*)ws; ws += (size_t)BB * SS * EE * 2;    // 32 MB
  __bf16* Vtb    = (__bf16*)ws; ws += (size_t)BB * SS * EE * 2;    // 32 MB
  __bf16* ctxb   = (__bf16*)ws;                                    // 32 MB

  cvt_f32_bf16<<<(BB * SS * EE / 4 + 255) / 256, 256, 0, stream>>>(hidden, hid_bf,
                                                                   BB * SS * EE / 4);
  transpose_cvt<<<dim3(3 * EE / 32, EE / 32), dim3(32, 8), 0, stream>>>(qkv_w, qkv_wt,
                                                                        EE, 3 * EE);
  transpose_cvt<<<dim3(EE / 32, EE / 32), dim3(32, 8), 0, stream>>>(out_w, out_wt,
                                                                    EE, EE);
  gemm_bf16_wmma<__bf16><<<dim3(3 * EE / 64, BB * SS / 128), 256, 0, stream>>>(
      hid_bf, qkv_wt, qkv_b, qkv_bf, BB * SS, 3 * EE, EE);
  rope_split<<<dim3(SS, HH, BB), DD, 0, stream>>>(qkv_bf, Qb, Kb, Vtb);
  flash_attn_wmma<<<dim3(SS / 64, HH, BB), 128, 0, stream>>>(Qb, Kb, Vtb, ctxb);
  gemm_bf16_wmma<float><<<dim3(EE / 64, BB * SS / 128), 256, 0, stream>>>(
      ctxb, out_wt, out_b, out, BB * SS, EE, EE);
}
